// RegistrationRegression_13056700580599
// MI455X (gfx1250) — matrix operand order repaired and checked by reading, not compiled, not run on hardware
//
#include <hip/hip_runtime.h>
#include <math.h>

#define BB 8
#define NN 2048
#define KF 1024

typedef __attribute__((ext_vector_type(16))) _Float16 v16h;
typedef __attribute__((ext_vector_type(8)))  float    v8f;

union AV { v16h v; uint4 q[2]; };
union CV { v8f  v; float f[8]; };

__device__ inline v8f zero_v8f() {
  v8f z = {0.f, 0.f, 0.f, 0.f, 0.f, 0.f, 0.f, 0.f};
  return z;
}

// ---- WMMA helpers (16x16x32 f16 -> f32) -------------------------------------

__device__ inline v8f wmma_f16(v16h a, v16h b, v8f c) {
  return __builtin_amdgcn_wmma_f32_16x16x32_f16(false, a, false, b, (short)0, c,
                                                false, false);
}

// A tile stored pre-swizzled in WMMA lane order: 32 lanes x 16 contiguous
// halves (32B per lane) -> 2x b128 loads.
__device__ inline v16h load_a_swz(const _Float16* tile, int lane) {
  AV a;
  const uint4* p = (const uint4*)(tile + lane * 16);
  a.q[0] = p[0];
  a.q[1] = p[1];
  return a.v;
}

// B tile from pre-swizzled weights: lane holds 16 contiguous halves.
__device__ inline v16h load_b_tile(const _Float16* wsw, int ntiles, int kt,
                                   int nt, int lane) {
  AV b;
  const uint4* p =
      (const uint4*)(wsw + ((size_t)(kt * ntiles + nt) * 32 + lane) * 16);
  b.q[0] = p[0];
  b.q[1] = p[1];
  return b.v;
}

// Map element coords (Mlocal in 0..15, Koff in 0..31) of a 16x32 f16 A tile to
// (lane, half) per ISA layout: lane l, half j: M=l&15,
// K = j + (j<8 ? (l<16?0:8) : (l<16?8:16))
__device__ inline void a_swz_coord(int Mlocal, int Koff, int& lp, int& j) {
  int g = Koff >> 3;  // 0..3
  lp = Mlocal + ((g & 1) ? 16 : 0);
  j = (Koff & 7) + ((g >= 2) ? 8 : 0);
}

// Async global->LDS copy of one dword (gfx1250, tracked by ASYNCcnt).
__device__ inline void async_copy_dw(const float* gsrc, float* lds_dst) {
  unsigned loff = (unsigned)(size_t)lds_dst;
  unsigned long long ga = (unsigned long long)(size_t)gsrc;
  asm volatile("global_load_async_to_lds_b32 %0, %1, off" ::"v"(loff), "v"(ga)
               : "memory");
}
__device__ inline void async_wait0() {
  asm volatile("s_wait_asynccnt 0" ::: "memory");
}

// ---- SE(3) exp --------------------------------------------------------------

__device__ inline void se3_exp6(const float* x, float* G) {
  float wx = x[0], wy = x[1], wz = x[2];
  float vx = x[3], vy = x[4], vz = x[5];
  float t2 = wx * wx + wy * wy + wz * wz;
  float t = sqrtf(t2);
  float a, b, c;
  if (t < 1e-6f) {
    a = 1.f - t2 / 6.f;
    b = 0.5f - t2 / 24.f;
    c = 1.f / 6.f - t2 / 120.f;
  } else {
    float s = sinf(t), co = cosf(t);
    a = s / t;
    b = (1.f - co) / t2;
    c = (t - s) / (t2 * t);
  }
  float W[9] = {0.f, -wz, wy, wz, 0.f, -wx, -wy, wx, 0.f};
  float W2[9];
#pragma unroll
  for (int i = 0; i < 3; ++i)
#pragma unroll
    for (int j = 0; j < 3; ++j) {
      float s = 0.f;
#pragma unroll
      for (int k = 0; k < 3; ++k) s += W[i * 3 + k] * W[k * 3 + j];
      W2[i * 3 + j] = s;
    }
  float R[9], V[9];
#pragma unroll
  for (int i = 0; i < 9; ++i) {
    float id = (i % 4 == 0) ? 1.f : 0.f;
    R[i] = id + a * W[i] + b * W2[i];
    V[i] = id + b * W[i] + c * W2[i];
  }
  float px = V[0] * vx + V[1] * vy + V[2] * vz;
  float py = V[3] * vx + V[4] * vy + V[5] * vz;
  float pz = V[6] * vx + V[7] * vy + V[8] * vz;
  G[0] = R[0]; G[1] = R[1]; G[2]  = R[2]; G[3]  = px;
  G[4] = R[3]; G[5] = R[4]; G[6]  = R[5]; G[7]  = py;
  G[8] = R[6]; G[9] = R[7]; G[10] = R[8]; G[11] = pz;
  G[12] = 0.f; G[13] = 0.f; G[14] = 0.f; G[15] = 1.f;
}

// ---- Small kernels ----------------------------------------------------------

__global__ __launch_bounds__(256) void k_means(const float* pts, float* mean) {
  __shared__ float sm[3][256];
  int b = blockIdx.x, t = threadIdx.x;
  float ax = 0.f, ay = 0.f, az = 0.f;
  for (int n = t; n < NN; n += 256) {
    const float* p = pts + ((size_t)b * NN + n) * 3;
    ax += p[0]; ay += p[1]; az += p[2];
  }
  sm[0][t] = ax; sm[1][t] = ay; sm[2][t] = az;
  __syncthreads();
  for (int o = 128; o > 0; o >>= 1) {
    if (t < o) {
      sm[0][t] += sm[0][t + o];
      sm[1][t] += sm[1][t + o];
      sm[2][t] += sm[2][t + o];
    }
    __syncthreads();
  }
  if (t < 3) mean[b * 3 + t] = sm[t][0] * (1.f / (float)NN);
}

__global__ void k_center(const float* p0, const float* p1, const float* p0m,
                         const float* p1m, float* cb) {
  int idx = blockIdx.x * 256 + threadIdx.x;
  if (idx >= 16 * NN) return;
  int c = idx >> 11, n = idx & (NN - 1);
  int sel = c >> 3, b = c & 7;
  const float* src = sel ? p1 : p0;
  const float* m = sel ? p1m : p0m;
#pragma unroll
  for (int d = 0; d < 3; ++d)
    cb[(size_t)c * NN * 3 + n * 3 + d] =
        src[((size_t)b * NN + n) * 3 + d] - m[b * 3 + d];
}

__global__ void k_dmat(const float* dt, float* Dm) {
  int j = threadIdx.x;
  if (j >= 6) return;
  float x[6] = {0.f, 0.f, 0.f, 0.f, 0.f, 0.f};
  x[j] = -dt[j];
  se3_exp6(x, Dm + j * 16);
}

__global__ void k_pj(const float* cb_in, const float* Dm, float* cb) {
  int idx = blockIdx.x * 256 + threadIdx.x;
  if (idx >= 48 * NN) return;
  int s = idx >> 11, n = idx & (NN - 1);
  int b = s / 6, j = s % 6;
  const float* G = Dm + j * 16;
  const float* q = cb_in + (size_t)b * NN * 3 + n * 3;
  float x = q[0], y = q[1], z = q[2];
  float* o = cb + (size_t)(16 + s) * NN * 3 + n * 3;
  o[0] = G[0] * x + G[1] * y + G[2] * z + G[3];
  o[1] = G[4] * x + G[5] * y + G[6] * z + G[7];
  o[2] = G[8] * x + G[9] * y + G[10] * z + G[11];
}

// Convert row-major f32 weight [Kd][Nd] into WMMA-B swizzled f16 tiles.
__global__ void k_cvtw(const float* W, _Float16* out, int Kd, int Nd) {
  int idx = blockIdx.x * 256 + threadIdx.x;
  if (idx >= Kd * Nd) return;
  int j = idx & 15;
  int l = (idx >> 4) & 31;
  int tt = idx >> 9;
  int NT = Nd >> 4;
  int nt = tt % NT;
  int kt = tt / NT;
  int k = kt * 32 + ((j < 8) ? j : j + 8) + ((l < 16) ? 0 : 8);
  int n = nt * 16 + (l & 15);
  out[idx] = (_Float16)W[(size_t)k * Nd + n];
}

// Convert row-major f32 activations [16][1024] into WMMA-A swizzled f16 tiles.
__global__ void k_cvtA(const float* in, _Float16* out) {
  int o = blockIdx.x * 256 + threadIdx.x;
  if (o >= 16 * KF) return;
  int j = o & 15;
  int lp = (o >> 4) & 31;
  int kt = o >> 9;
  int Ml = lp & 15;
  int hi = (lp < 16) ? 0 : 8;
  int Koff = (j < 8) ? (j + hi) : (j + 8 + hi);
  out[o] = (_Float16)in[Ml * KF + kt * 32 + Koff];
}

// ---- Encoder: per-cloud MLP (3->64->128->1024) + maxpool --------------------

__global__ __launch_bounds__(256) void k_encoder(
    const float* pts, const float* ew1, const float* eb1, const _Float16* w2s,
    const float* eb2, const _Float16* w3s, const float* eb3, float* feats) {
  __shared__ float pl[128][3];
  __shared__ _Float16 h1s[8][2][32][16];  // A-swizzled [mt][kt][lane][half]
  __shared__ _Float16 h2s[8][4][32][16];  // A-swizzled [mt][kt][lane][half]
  __shared__ float w1s[192];
  __shared__ float b1s[64];
  __shared__ float b2s[128];
  __shared__ float b3s[1024];
  int cloud = blockIdx.x;
  int t = threadIdx.x, lane = t & 31, wv = t >> 5;
  if (t < 192) w1s[t] = ew1[t];
  if (t < 64) b1s[t] = eb1[t];
  if (t < 128) b2s[t] = eb2[t];
  for (int i = t; i < 1024; i += 256) b3s[i] = eb3[i];
  float wmax[8];
#pragma unroll
  for (int q = 0; q < 8; ++q) wmax[q] = -3.4e38f;
  const float* base = pts + (size_t)cloud * NN * 3;

  for (int pt = 0; pt < 16; ++pt) {
    __syncthreads();
    // async-stage the 128-point tile into LDS (ASYNCcnt path)
    for (int i = t; i < 384; i += 256)
      async_copy_dw(base + pt * 384 + i, &pl[0][0] + i);
    async_wait0();
    __syncthreads();
    // layer 1 (K=3): plain VALU, store A-swizzled f16
    for (int i = t; i < 128 * 64; i += 256) {
      int m = i >> 6, cc = i & 63;
      float h = b1s[cc] + pl[m][0] * w1s[cc] + pl[m][1] * w1s[64 + cc] +
                pl[m][2] * w1s[128 + cc];
      int lp, j;
      a_swz_coord(m & 15, cc & 31, lp, j);
      h1s[m >> 4][cc >> 5][lp][j] = (_Float16)fmaxf(h, 0.f);
    }
    __syncthreads();
    // layer 2: [128,64]x[64,128] WMMA; wave wv owns m-tile wv
    {
      int mt = wv;
      v16h a0 = load_a_swz(&h1s[mt][0][0][0], lane);
      v16h a1 = load_a_swz(&h1s[mt][1][0][0], lane);
      for (int nt = 0; nt < 8; ++nt) {
        CV acc;
        acc.v = zero_v8f();
        acc.v = wmma_f16(a0, load_b_tile(w2s, 8, 0, nt, lane), acc.v);
        acc.v = wmma_f16(a1, load_b_tile(w2s, 8, 1, nt, lane), acc.v);
        int Nc = nt * 16 + (lane & 15);
        float bv = b2s[Nc];
        int mbase = (lane < 16) ? 0 : 8;
        int lp, j;
        a_swz_coord(0, Nc & 31, lp, j);  // lp base for Mlocal=0
        int kt = Nc >> 5;
#pragma unroll
        for (int r = 0; r < 8; ++r)
          h2s[mt][kt][lp + mbase + r][j] =
              (_Float16)fmaxf(acc.f[r] + bv, 0.f);
      }
    }
    __syncthreads();
    // layer 3: [128,128]x[128,1024] WMMA; A tiles hoisted, B streamed from L2
    for (int mt = 0; mt < 8; ++mt) {
      v16h at[4];
#pragma unroll
      for (int kt = 0; kt < 4; ++kt)
        at[kt] = load_a_swz(&h2s[mt][kt][0][0], lane);
      for (int q = 0; q < 8; ++q) {
        int gnt = wv * 8 + q;
        __builtin_prefetch(w3s + (size_t)gnt * 32 * 16, 0, 1);
        CV acc;
        acc.v = zero_v8f();
#pragma unroll
        for (int kt = 0; kt < 4; ++kt)
          acc.v = wmma_f16(at[kt], load_b_tile(w3s, 64, kt, gnt, lane), acc.v);
        float tm = acc.f[0];
#pragma unroll
        for (int r = 1; r < 8; ++r) tm = fmaxf(tm, acc.f[r]);
        tm = fmaxf(tm + b3s[gnt * 16 + (lane & 15)], 0.f);
        wmax[q] = fmaxf(wmax[q], tm);
      }
    }
  }
#pragma unroll
  for (int q = 0; q < 8; ++q) {
    float o = __shfl_xor(wmax[q], 16);
    float m = fmaxf(wmax[q], o);
    if (lane < 16)
      feats[(size_t)cloud * KF + (wv * 8 + q) * 16 + lane] = m;
  }
}

// ---- Decoder: [16,1024]x[1024,1024] relu, then x[1024,6144] -----------------

__global__ __launch_bounds__(256) void k_decoder(const _Float16* xf16s,
                                                 const _Float16* dw1s,
                                                 const float* db1,
                                                 const _Float16* dw2s,
                                                 const float* db2, float* y01) {
  __shared__ _Float16 Hs[32][32][16];  // A-swizzled [kt][lane][half]
  int t = threadIdx.x, lane = t & 31, wv = t >> 5;
  for (int q = 0; q < 8; ++q) {
    int nt = wv * 8 + q;
    CV acc;
    acc.v = zero_v8f();
    for (int kt = 0; kt < 32; ++kt) {
      v16h a = load_a_swz(xf16s + kt * 512, lane);
      acc.v = wmma_f16(a, load_b_tile(dw1s, 64, kt, nt, lane), acc.v);
    }
    int Nc = nt * 16 + (lane & 15);
    float bv = db1[Nc];
    int mbase = (lane < 16) ? 0 : 8;
    int lp, j;
    a_swz_coord(0, Nc & 31, lp, j);
    int kt2 = Nc >> 5;
#pragma unroll
    for (int r = 0; r < 8; ++r)
      Hs[kt2][lp + mbase + r][j] = (_Float16)fmaxf(acc.f[r] + bv, 0.f);
  }
  __syncthreads();
  for (int q = 0; q < 48; ++q) {
    int nt = wv * 48 + q;
    CV acc;
    acc.v = zero_v8f();
    for (int kt = 0; kt < 32; ++kt) {
      v16h a = load_a_swz(&Hs[kt][0][0], lane);
      acc.v = wmma_f16(a, load_b_tile(dw2s, 384, kt, nt, lane), acc.v);
    }
    float bv = db2[nt * 16 + (lane & 15)];
    int mrow = (lane < 16) ? 0 : 8;
    int ncol = nt * 16 + (lane & 15);
#pragma unroll
    for (int r = 0; r < 8; ++r)
      y01[(size_t)(mrow + r) * 6144 + ncol] = acc.f[r] + bv;
  }
}

// ---- Chamfer: for each row point of x, min squared distance to y ------------

__global__ __launch_bounds__(128) void k_chamfer(const float* xb, size_t xstr,
                                                 const float* yb, size_t ystr,
                                                 float* outmin) {
  int b = blockIdx.x;
  int n = blockIdx.y * 128 + threadIdx.x;
  const float* x = xb + (size_t)b * xstr;
  const float* y = yb + (size_t)b * ystr;
  float px = x[n * 3 + 0], py = x[n * 3 + 1], pz = x[n * 3 + 2];
  float rx = px * px + py * py + pz * pz;
  __shared__ float ys[128][3];
  float best = 3.4e38f;
  for (int mt = 0; mt < 16; ++mt) {
    __syncthreads();
    for (int i = threadIdx.x; i < 384; i += 128)
      async_copy_dw(y + mt * 384 + i, &ys[0][0] + i);
    async_wait0();
    __syncthreads();
    for (int m = 0; m < 128; ++m) {
      float qx = ys[m][0], qy = ys[m][1], qz = ys[m][2];
      float ry = qx * qx + qy * qy + qz * qz;
      float d = rx + ry - 2.f * (px * qx + py * qy + pz * qz);
      best = fminf(best, d);
    }
  }
  outmin[b * NN + n] = best;
}

__global__ __launch_bounds__(256) void k_lossreduce(const float* cham,
                                                    float* loss) {
  __shared__ float sm[256];
  int t = threadIdx.x;
  float s = 0.f;
  for (int i = t; i < 4 * BB * NN; i += 256) s += cham[i];
  sm[t] = s;
  __syncthreads();
  for (int o = 128; o > 0; o >>= 1) {
    if (t < o) sm[t] += sm[t + o];
    __syncthreads();
  }
  if (t == 0) loss[0] = sm[0] / (float)(BB * NN);
}

// ---- Jacobian / Gauss-Newton pieces ----------------------------------------

__global__ void k_jt(const float* feats, const float* dt, float* Jt) {
  int idx = blockIdx.x * 256 + threadIdx.x;
  if (idx >= BB * 6 * KF) return;
  int b = idx / (6 * KF);
  int r = idx % (6 * KF);
  int i = r / KF, k = r % KF;
  Jt[idx] =
      (feats[(size_t)b * KF + k] - feats[(size_t)(16 + b * 6 + i) * KF + k]) /
      dt[i];
}

__global__ __launch_bounds__(256) void k_h(const float* Jt, float* H) {
  int b = blockIdx.x;
  int lane = threadIdx.x & 31, wv = threadIdx.x >> 5;
  const float* J = Jt + (size_t)b * 6 * KF;
  for (int p = wv; p < 36; p += 8) {
    int i = p / 6, j = p % 6;
    float s = 0.f;
    for (int k = lane; k < KF; k += 32) s += J[i * KF + k] * J[j * KF + k];
    for (int o = 16; o > 0; o >>= 1) s += __shfl_xor(s, o);
    if (lane == 0) H[b * 36 + p] = s;
  }
}

__global__ void k_inv6(const float* H, float* Hinv) {
  int b = threadIdx.x;
  if (b >= BB) return;
  float A[6][6], M[6][6];
  for (int i = 0; i < 6; ++i)
    for (int j = 0; j < 6; ++j) {
      A[i][j] = H[b * 36 + i * 6 + j];
      M[i][j] = (i == j) ? 1.f : 0.f;
    }
  for (int c = 0; c < 6; ++c) {
    int piv = c;
    float mx = fabsf(A[c][c]);
    for (int r = c + 1; r < 6; ++r) {
      float v = fabsf(A[r][c]);
      if (v > mx) { mx = v; piv = r; }
    }
    if (piv != c)
      for (int k = 0; k < 6; ++k) {
        float ta = A[c][k]; A[c][k] = A[piv][k]; A[piv][k] = ta;
        float tm = M[c][k]; M[c][k] = M[piv][k]; M[piv][k] = tm;
      }
    float d = 1.f / A[c][c];
    for (int k = 0; k < 6; ++k) { A[c][k] *= d; M[c][k] *= d; }
    for (int r = 0; r < 6; ++r)
      if (r != c) {
        float f = A[r][c];
        for (int k = 0; k < 6; ++k) {
          A[r][k] -= f * A[c][k];
          M[r][k] -= f * M[c][k];
        }
      }
  }
  for (int i = 0; i < 36; ++i) Hinv[b * 36 + i] = M[i / 6][i % 6];
}

__global__ void k_pinv(const float* Hinv, const float* Jt, float* P) {
  int idx = blockIdx.x * 256 + threadIdx.x;
  if (idx >= BB * 6 * KF) return;
  int b = idx / (6 * KF);
  int r = idx % (6 * KF);
  int i = r / KF, k = r % KF;
  float s = 0.f;
#pragma unroll
  for (int j = 0; j < 6; ++j)
    s += Hinv[b * 36 + i * 6 + j] * Jt[(size_t)b * 6 * KF + j * KF + k];
  P[idx] = s;
}

__global__ void k_init(float* g, float* r, float* flags) {
  int idx = blockIdx.x * 256 + threadIdx.x;
  if (idx < BB * KF) r[idx] = 0.f;
  if (idx < 128) g[idx] = (((idx & 15) % 5) == 0) ? 1.f : 0.f;
  if (idx == 0) flags[0] = 0.f;
}

__global__ void k_transform(const float* g, const float* q1, float* out) {
  int idx = blockIdx.x * 256 + threadIdx.x;
  if (idx >= BB * NN) return;
  int b = idx >> 11, n = idx & (NN - 1);
  const float* G = g + b * 16;
  const float* q = q1 + (size_t)b * NN * 3 + n * 3;
  float x = q[0], y = q[1], z = q[2];
  float* o = out + (size_t)b * NN * 3 + n * 3;
  o[0] = G[0] * x + G[1] * y + G[2] * z + G[3];
  o[1] = G[4] * x + G[5] * y + G[6] * z + G[7];
  o[2] = G[8] * x + G[9] * y + G[10] * z + G[11];
}

__global__ __launch_bounds__(256) void k_update(const float* fiter,
                                                const float* f0,
                                                const float* P, float* r,
                                                float* g, float* flags) {
  __shared__ float dx[BB][6];
  __shared__ float nrm[BB];
  __shared__ float ctl[2];
  int t = threadIdx.x, lane = t & 31, wv = t >> 5;
  for (int i = 0; i < 6; ++i) {
    float s = 0.f;
    for (int k = lane; k < KF; k += 32) {
      float rk = fiter[wv * KF + k] - f0[wv * KF + k];
      s += P[(size_t)(wv * 6 + i) * KF + k] * rk;
    }
    for (int o = 16; o > 0; o >>= 1) s += __shfl_xor(s, o);
    if (lane == 0) dx[wv][i] = -s;
  }
  __syncthreads();
  if (t < BB) {
    float s = 0.f;
    for (int i = 0; i < 6; ++i) s += dx[t][i] * dx[t][i];
    nrm[t] = sqrtf(s);
  }
  __syncthreads();
  if (t == 0) {
    float check = 0.f;
    for (int b = 0; b < BB; ++b) check = fmaxf(check, nrm[b]);
    float done = flags[0];
    ctl[0] = done;
    float stop = (done != 0.f || check < 1e-7f) ? 1.f : 0.f;
    ctl[1] = stop;
    flags[0] = stop;
  }
  __syncthreads();
  float old_done = ctl[0], stop = ctl[1];
  if (old_done == 0.f)
    for (int idx = t; idx < BB * KF; idx += 256)
      r[idx] = fiter[idx] - f0[idx];
  if (stop == 0.f && t < BB) {
    float G[16];
    se3_exp6(dx[t], G);
    float go[16], gn[16];
    for (int i = 0; i < 16; ++i) go[i] = g[t * 16 + i];
    for (int i = 0; i < 4; ++i)
      for (int j = 0; j < 4; ++j) {
        float s = 0.f;
        for (int u = 0; u < 4; ++u) s += G[i * 4 + u] * go[u * 4 + j];
        gn[i * 4 + j] = s;
      }
    for (int i = 0; i < 16; ++i) g[t * 16 + i] = gn[i];
  }
}

__global__ void k_final(const float* r, const float* loss, const float* g,
                        const float* p0m, const float* p1m, float* out) {
  int idx = blockIdx.x * 256 + threadIdx.x;
  if (idx < BB * KF) out[idx] = r[idx];
  else if (idx == BB * KF) out[idx] = loss[0];
  else if (idx < BB * KF + 1 + 128) {
    int e = idx - (BB * KF + 1);
    int b = e / 16, i = (e % 16) / 4, j = e % 4;
    float s = 0.f;
    for (int tt = 0; tt < 4; ++tt) {
      float a0v = (i == tt ? 1.f : 0.f) +
                  ((tt == 3 && i < 3) ? p0m[b * 3 + i] : 0.f);
      float m1 = 0.f;
      for (int u = 0; u < 4; ++u) {
        float a1v = (u == j ? 1.f : 0.f) +
                    ((u < 3 && j == 3) ? -p1m[b * 3 + u] : 0.f);
        m1 += g[b * 16 + tt * 4 + u] * a1v;
      }
      s += a0v * m1;
    }
    out[idx] = s;
  }
}

// ---- Host launch ------------------------------------------------------------

extern "C" void kernel_launch(void* const* d_in, const int* in_sizes, int n_in,
                              void* d_out, int out_size, void* d_ws,
                              size_t ws_size, hipStream_t stream) {
  const float* p0  = (const float*)d_in[0];
  const float* p1  = (const float*)d_in[1];
  const float* dt  = (const float*)d_in[2];
  const float* ew1 = (const float*)d_in[3];
  const float* eb1 = (const float*)d_in[4];
  const float* ew2 = (const float*)d_in[5];
  const float* eb2 = (const float*)d_in[6];
  const float* ew3 = (const float*)d_in[7];
  const float* eb3 = (const float*)d_in[8];
  const float* dw1 = (const float*)d_in[9];
  const float* db1 = (const float*)d_in[10];
  const float* dw2 = (const float*)d_in[11];
  const float* db2 = (const float*)d_in[12];

  char* ws = (char*)d_ws;
  size_t off = 0;
  auto alloc = [&](size_t bytes) -> void* {
    void* p = ws + off;
    off += (bytes + 255) & ~(size_t)255;
    return p;
  };
  float* cloudbuf = (float*)alloc((size_t)64 * NN * 3 * 4);  // q0|q1|pJ
  float* ptrans   = (float*)alloc((size_t)BB * NN * 3 * 4);
  float* featbuf  = (float*)alloc((size_t)64 * KF * 4);      // f0|f1|fJ
  float* fiter    = (float*)alloc((size_t)BB * KF * 4);
  float* y01      = (float*)alloc((size_t)16 * 6144 * 4);
  float* cham     = (float*)alloc((size_t)4 * BB * NN * 4);
  float* Jt       = (float*)alloc((size_t)BB * 6 * KF * 4);
  float* Pinv     = (float*)alloc((size_t)BB * 6 * KF * 4);
  float* Hb       = (float*)alloc(288 * 4);
  float* Hinv     = (float*)alloc(288 * 4);
  float* gbuf     = (float*)alloc(128 * 4);
  float* rbuf     = (float*)alloc((size_t)BB * KF * 4);
  float* flags    = (float*)alloc(16 * 4);
  float* p0m      = (float*)alloc(24 * 4);
  float* p1m      = (float*)alloc(24 * 4);
  float* Dm       = (float*)alloc(96 * 4);
  float* lossbuf  = (float*)alloc(16 * 4);
  _Float16* xf16s = (_Float16*)alloc((size_t)16 * KF * 2);
  _Float16* w2s   = (_Float16*)alloc((size_t)64 * 128 * 2);
  _Float16* w3s   = (_Float16*)alloc((size_t)128 * KF * 2);
  _Float16* dw1s  = (_Float16*)alloc((size_t)KF * 1024 * 2);
  _Float16* dw2s  = (_Float16*)alloc((size_t)KF * 6144 * 2);

  k_means<<<BB, 256, 0, stream>>>(p0, p0m);
  k_means<<<BB, 256, 0, stream>>>(p1, p1m);
  k_center<<<(16 * NN + 255) / 256, 256, 0, stream>>>(p0, p1, p0m, p1m,
                                                      cloudbuf);
  k_dmat<<<1, 32, 0, stream>>>(dt, Dm);
  k_pj<<<(48 * NN + 255) / 256, 256, 0, stream>>>(cloudbuf, Dm, cloudbuf);

  k_cvtw<<<(64 * 128 + 255) / 256, 256, 0, stream>>>(ew2, w2s, 64, 128);
  k_cvtw<<<(128 * KF + 255) / 256, 256, 0, stream>>>(ew3, w3s, 128, KF);
  k_cvtw<<<(KF * 1024 + 255) / 256, 256, 0, stream>>>(dw1, dw1s, KF, 1024);
  k_cvtw<<<(KF * 6144 + 255) / 256, 256, 0, stream>>>(dw2, dw2s, KF, 6144);

  k_encoder<<<64, 256, 0, stream>>>(cloudbuf, ew1, eb1, w2s, eb2, w3s, eb3,
                                    featbuf);

  k_cvtA<<<(16 * KF + 255) / 256, 256, 0, stream>>>(featbuf, xf16s);
  k_decoder<<<1, 256, 0, stream>>>(xf16s, dw1s, db1, dw2s, db2, y01);

  k_chamfer<<<dim3(BB, 16), 128, 0, stream>>>(y01, 6144, cloudbuf, 6144,
                                              cham + 0 * BB * NN);
  k_chamfer<<<dim3(BB, 16), 128, 0, stream>>>(cloudbuf, 6144, y01, 6144,
                                              cham + 1 * BB * NN);
  k_chamfer<<<dim3(BB, 16), 128, 0, stream>>>(y01 + (size_t)8 * 6144, 6144,
                                              cloudbuf + (size_t)8 * 6144, 6144,
                                              cham + 2 * BB * NN);
  k_chamfer<<<dim3(BB, 16), 128, 0, stream>>>(cloudbuf + (size_t)8 * 6144, 6144,
                                              y01 + (size_t)8 * 6144, 6144,
                                              cham + 3 * BB * NN);
  k_lossreduce<<<1, 256, 0, stream>>>(cham, lossbuf);

  k_jt<<<(BB * 6 * KF + 255) / 256, 256, 0, stream>>>(featbuf, dt, Jt);
  k_h<<<BB, 256, 0, stream>>>(Jt, Hb);
  k_inv6<<<1, 32, 0, stream>>>(Hb, Hinv);
  k_pinv<<<(BB * 6 * KF + 255) / 256, 256, 0, stream>>>(Hinv, Jt, Pinv);

  k_init<<<(BB * KF + 255) / 256, 256, 0, stream>>>(gbuf, rbuf, flags);
  for (int it = 0; it < 5; ++it) {
    k_transform<<<(BB * NN + 255) / 256, 256, 0, stream>>>(
        gbuf, cloudbuf + (size_t)8 * NN * 3, ptrans);
    k_encoder<<<BB, 256, 0, stream>>>(ptrans, ew1, eb1, w2s, eb2, w3s, eb3,
                                      fiter);
    k_update<<<1, 256, 0, stream>>>(fiter, featbuf, Pinv, rbuf, gbuf, flags);
  }

  k_final<<<(BB * KF + 1 + 128 + 255) / 256, 256, 0, stream>>>(
      rbuf, lossbuf, gbuf, p0m, p1m, (float*)d_out);
  (void)in_sizes; (void)n_in; (void)out_size; (void)ws_size;
}
